// AxialAttention_76020921139692
// MI455X (gfx1250) — matrix operand order repaired and checked by reading, not compiled, not run on hardware
//
#include <hip/hip_runtime.h>
#include <hip/hip_bf16.h>
#include <math.h>

typedef _Float16 v16h __attribute__((ext_vector_type(16)));
typedef float    v8f  __attribute__((ext_vector_type(8)));

#define N_TOK   32768
#define D_MODEL 256
#define INNER_D 512
#define N_HEADS 8
#define DIM_HEAD 64
#define SEQ     512
#define NBATCH  64

union F16x16 { v16h v; uint4 q[2]; _Float16 h[16]; };

__device__ __forceinline__ v8f wmma16(v16h a, v16h b, v8f c) {
  // D = A(16x32 f16) * B(32x16 f16) + C(16x16 f32)
  return __builtin_amdgcn_wmma_f32_16x16x32_f16(false, a, false, b, (short)0, c, false, false);
}

// reductions across the 16-lane half-wave that shares one matrix row (wave32)
__device__ __forceinline__ float red_sum16(float v) {
  v += __shfl_xor(v, 1, 32); v += __shfl_xor(v, 2, 32);
  v += __shfl_xor(v, 4, 32); v += __shfl_xor(v, 8, 32);
  return v;
}
__device__ __forceinline__ float red_max16(float v) {
  v = fmaxf(v, __shfl_xor(v, 1, 32)); v = fmaxf(v, __shfl_xor(v, 2, 32));
  v = fmaxf(v, __shfl_xor(v, 4, 32)); v = fmaxf(v, __shfl_xor(v, 8, 32));
  return v;
}

// ---------------- 1. LayerNorm + cast to f16 ----------------
__global__ void ln_f16_kernel(const float* __restrict__ x, const float* __restrict__ g,
                              const float* __restrict__ be, _Float16* __restrict__ xn) {
  const int row = blockIdx.x;
  const int l = threadIdx.x;
  const float* xr = x + (size_t)row * D_MODEL + l * 8;
  float4 a = *(const float4*)xr;
  float4 c = *(const float4*)(xr + 4);
  float xv[8] = {a.x, a.y, a.z, a.w, c.x, c.y, c.z, c.w};
  float s = 0.f, s2 = 0.f;
  #pragma unroll
  for (int k = 0; k < 8; ++k) { s += xv[k]; s2 += xv[k] * xv[k]; }
  #pragma unroll
  for (int m = 1; m < 32; m <<= 1) { s += __shfl_xor(s, m, 32); s2 += __shfl_xor(s2, m, 32); }
  const float mean = s * (1.0f / 256.0f);
  const float var  = s2 * (1.0f / 256.0f) - mean * mean;
  const float rstd = rsqrtf(var + 1e-5f);
  const float* gp = g + l * 8;
  const float* bp = be + l * 8;
  union { uint4 q; _Float16 h[8]; } o;
  #pragma unroll
  for (int k = 0; k < 8; ++k)
    o.h[k] = (_Float16)(gp[k] * (xv[k] - mean) * rstd + bp[k]);
  *(uint4*)(xn + (size_t)row * D_MODEL + l * 8) = o.q;
}

// ---------------- 2. Weight convert/transpose to f16 [out][in] ----------------
__global__ void cvt_wqkvg_kernel(const float* __restrict__ Wq, const float* __restrict__ Wkv,
                                 const float* __restrict__ Wg, _Float16* __restrict__ WT) {
  const int o = blockIdx.x;   // 0..2047  (Q:0-511, K:512-1023, V:1024-1535, G:1536-2047)
  const int i = threadIdx.x;  // 0..255
  float v;
  if (o < 512)       v = Wq [(size_t)i * 512  + o];
  else if (o < 1536) v = Wkv[(size_t)i * 1024 + (o - 512)];
  else               v = Wg [(size_t)i * 512  + (o - 1536)];
  WT[(size_t)o * D_MODEL + i] = (_Float16)v;
}

__global__ void cvt_wo_kernel(const float* __restrict__ Wo, _Float16* __restrict__ WoT) {
  const int o = blockIdx.x;   // 0..255
  for (int i = threadIdx.x; i < INNER_D; i += 256)
    WoT[(size_t)o * INNER_D + i] = (_Float16)Wo[(size_t)i * D_MODEL + o];
}

// ---------------- 3. Fused Q/K/V/gate projection GEMM ----------------
// one wave computes a 16-token x 128-out tile; A fragment reused over 8 out tiles.
// All fragment loads are immediate offsets from two per-lane base pointers.
__global__ void proj_kernel(const _Float16* __restrict__ XN, const _Float16* __restrict__ WT,
                            const float* __restrict__ bg,
                            _Float16* __restrict__ Qb, _Float16* __restrict__ Kb,
                            _Float16* __restrict__ Vt, _Float16* __restrict__ Gb) {
  const int l = threadIdx.x, lr = l & 15, lh = l >> 4;
  const int tt = blockIdx.x;          // token tile (16 tokens)
  const int og = blockIdx.y;          // 128-wide out group
  v8f zero = {};
  v8f acc[8];
  #pragma unroll
  for (int i = 0; i < 8; ++i) acc[i] = zero;
  // A: lane holds row (tt*16+lr), halves kt*32 + lh*8 (+16)
  const _Float16* arow  = XN + (size_t)(tt * 16 + lr) * D_MODEL + lh * 8;
  // B: lane holds out column (og*128 + ot*16 + lr), halves kt*32 + lh*8 (+16)
  const _Float16* wbase = WT + (size_t)(og * 128 + lr) * D_MODEL + lh * 8;
  #pragma unroll
  for (int kt = 0; kt < D_MODEL / 32; ++kt) {
    F16x16 a;
    a.q[0] = *(const uint4*)(arow + kt * 32);
    a.q[1] = *(const uint4*)(arow + kt * 32 + 16);
    #pragma unroll
    for (int ot = 0; ot < 8; ++ot) {
      F16x16 b;
      b.q[0] = *(const uint4*)(wbase + ot * 16 * D_MODEL + kt * 32);
      b.q[1] = *(const uint4*)(wbase + ot * 16 * D_MODEL + kt * 32 + 16);
      acc[ot] = wmma16(a.v, b.v, acc[ot]);
    }
  }
  #pragma unroll
  for (int ot = 0; ot < 8; ++ot) {
    const int o = og * 128 + ot * 16 + lr;
    #pragma unroll
    for (int j = 0; j < 8; ++j) {
      const int t = tt * 16 + lh * 8 + j;
      const int bi = t >> 9, w = t & 511;
      const float val = acc[ot][j];
      if (o < 512) {                              // Q, pre-scaled by dh^-0.5
        const int h = o >> 6, dh = o & 63;
        Qb[(((size_t)(bi * 8 + h)) * SEQ + w) * DIM_HEAD + dh] = (_Float16)(val * 0.125f);
      } else if (o < 1024) {                      // K
        const int oo = o - 512, h = oo >> 6, dh = oo & 63;
        Kb[(((size_t)(bi * 8 + h)) * SEQ + w) * DIM_HEAD + dh] = (_Float16)val;
      } else if (o < 1536) {                      // V stored transposed [bh][dh][w]
        const int oo = o - 1024, h = oo >> 6, dh = oo & 63;
        Vt[(((size_t)(bi * 8 + h)) * DIM_HEAD + dh) * SEQ + w] = (_Float16)val;
      } else {                                    // sigmoid gate multiplier
        const int oo = o - 1536;
        const float z = val + bg[oo];
        Gb[(size_t)t * INNER_D + oo] = (_Float16)(1.0f / (1.0f + __expf(-z)));
      }
    }
  }
}

// ---------------- 4. Flash attention (per head-batch, 16-query tile) ----------------
__global__ void attn_kernel(const _Float16* __restrict__ Qb, const _Float16* __restrict__ Kb,
                            const _Float16* __restrict__ Vt, const _Float16* __restrict__ Gb,
                            _Float16* __restrict__ AG) {
  __shared__ _Float16 Pld[16 * 32];               // P tile staged for A-fragment reload
  const int l = threadIdx.x, lr = l & 15, lh = l >> 4;
  const int qt = blockIdx.x, bh = blockIdx.y;
  const _Float16* Qp = Qb + (size_t)bh * SEQ * DIM_HEAD;
  // per-lane base pointers; all tile loads below are immediate offsets / one add per jc
  const _Float16* kbase = Kb + (size_t)bh * SEQ * DIM_HEAD + (size_t)lr * DIM_HEAD + lh * 8;
  const _Float16* vbase = Vt + (size_t)bh * DIM_HEAD * SEQ + (size_t)lr * SEQ + lh * 8;

  F16x16 qa[2];
  {
    const _Float16* p = Qp + (size_t)(qt * 16 + lr) * DIM_HEAD + lh * 8;
    qa[0].q[0] = *(const uint4*)(p);
    qa[0].q[1] = *(const uint4*)(p + 16);
    qa[1].q[0] = *(const uint4*)(p + 32);
    qa[1].q[1] = *(const uint4*)(p + 48);
  }
  v8f zero = {};
  float m_i[8], l_i[8];
  v8f o_acc[4];
  #pragma unroll
  for (int j = 0; j < 8; ++j) { m_i[j] = -1e30f; l_i[j] = 0.0f; }
  #pragma unroll
  for (int dt = 0; dt < 4; ++dt) o_acc[dt] = zero;

  for (int jc = 0; jc < SEQ / 32; ++jc) {
    const _Float16* kj = kbase + jc * 32 * DIM_HEAD;   // advance by one 32-key chunk
    const _Float16* vj = vbase + jc * 32;
    if (jc + 1 < SEQ / 32) {                           // prefetch next chunk (global_prefetch_b8)
      __builtin_prefetch(kj + 32 * DIM_HEAD, 0, 0);
      __builtin_prefetch(vj + 32, 0, 0);
    }
    v8f sv[2];
    #pragma unroll
    for (int nt = 0; nt < 2; ++nt) {
      v8f s = zero;
      #pragma unroll
      for (int ks = 0; ks < 2; ++ks) {
        F16x16 kf;
        kf.q[0] = *(const uint4*)(kj + nt * 16 * DIM_HEAD + ks * 32);
        kf.q[1] = *(const uint4*)(kj + nt * 16 * DIM_HEAD + ks * 32 + 16);
        s = wmma16(qa[ks].v, kf.v, s);
      }
      sv[nt] = s;
    }
    float scale[8];
    #pragma unroll
    for (int j = 0; j < 8; ++j) {
      float t = red_max16(fmaxf(sv[0][j], sv[1][j]));
      const float mnew = fmaxf(m_i[j], t);
      scale[j] = __expf(m_i[j] - mnew);
      m_i[j] = mnew;
      const float pa = __expf(sv[0][j] - mnew);
      const float pb = __expf(sv[1][j] - mnew);
      const int row = lh * 8 + j;
      Pld[row * 32 + lr]      = (_Float16)pa;
      Pld[row * 32 + 16 + lr] = (_Float16)pb;
      l_i[j] = l_i[j] * scale[j] + red_sum16(pa + pb);
    }
    #pragma unroll
    for (int dt = 0; dt < 4; ++dt)
      #pragma unroll
      for (int j = 0; j < 8; ++j)
        o_acc[dt][j] *= scale[j];
    __syncthreads();
    F16x16 pf;
    {
      const _Float16* pp = &Pld[lr * 32 + lh * 8];
      pf.q[0] = *(const uint4*)pp;         // ds_load_b128
      pf.q[1] = *(const uint4*)(pp + 16);
    }
    __syncthreads();
    #pragma unroll
    for (int dt = 0; dt < 4; ++dt) {
      F16x16 vf;
      vf.q[0] = *(const uint4*)(vj + dt * 16 * SEQ);
      vf.q[1] = *(const uint4*)(vj + dt * 16 * SEQ + 16);
      o_acc[dt] = wmma16(pf.v, vf.v, o_acc[dt]);
    }
  }
  const int b = bh >> 3, h = bh & 7;
  #pragma unroll
  for (int j = 0; j < 8; ++j) {
    const int t = b * SEQ + qt * 16 + lh * 8 + j;
    const float inv_l = 1.0f / l_i[j];
    #pragma unroll
    for (int dt = 0; dt < 4; ++dt) {
      const int o = h * DIM_HEAD + dt * 16 + lr;
      const float val = o_acc[dt][j] * inv_l;
      const float gm = (float)Gb[(size_t)t * INNER_D + o];
      AG[(size_t)t * INNER_D + o] = (_Float16)(val * gm);
    }
  }
}

// ---------------- 5. Output projection GEMM + bias ----------------
__global__ void out_kernel(const _Float16* __restrict__ AG, const _Float16* __restrict__ WoT,
                           const float* __restrict__ bo, float* __restrict__ out) {
  const int l = threadIdx.x, lr = l & 15, lh = l >> 4;
  const int tt = blockIdx.x, og = blockIdx.y;   // og in {0,1}: 128 cols each
  v8f zero = {};
  v8f acc[8];
  #pragma unroll
  for (int i = 0; i < 8; ++i) acc[i] = zero;
  const _Float16* arow  = AG  + (size_t)(tt * 16 + lr) * INNER_D + lh * 8;
  const _Float16* wbase = WoT + (size_t)(og * 128 + lr) * INNER_D + lh * 8;
  for (int kt = 0; kt < INNER_D / 32; ++kt) {
    const _Float16* ak = arow  + kt * 32;
    const _Float16* wk = wbase + kt * 32;
    if (kt + 1 < INNER_D / 32) __builtin_prefetch(ak + 32, 0, 0);
    F16x16 a;
    a.q[0] = *(const uint4*)(ak);
    a.q[1] = *(const uint4*)(ak + 16);
    #pragma unroll
    for (int ot = 0; ot < 8; ++ot) {
      F16x16 bf;
      bf.q[0] = *(const uint4*)(wk + ot * 16 * INNER_D);
      bf.q[1] = *(const uint4*)(wk + ot * 16 * INNER_D + 16);
      acc[ot] = wmma16(a.v, bf.v, acc[ot]);
    }
  }
  #pragma unroll
  for (int ot = 0; ot < 8; ++ot) {
    const int o = og * 128 + ot * 16 + lr;
    const float bias = bo[o];
    #pragma unroll
    for (int j = 0; j < 8; ++j) {
      const int t = tt * 16 + lh * 8 + j;
      out[(size_t)t * D_MODEL + o] = acc[ot][j] + bias;
    }
  }
}

extern "C" void kernel_launch(void* const* d_in, const int* in_sizes, int n_in,
                              void* d_out, int out_size, void* d_ws, size_t ws_size,
                              hipStream_t stream) {
  (void)in_sizes; (void)n_in; (void)out_size; (void)ws_size;
  const float* x    = (const float*)d_in[0];
  const float* ln_g = (const float*)d_in[1];
  const float* ln_b = (const float*)d_in[2];
  const float* Wq   = (const float*)d_in[3];
  const float* Wkv  = (const float*)d_in[4];
  const float* Wg   = (const float*)d_in[5];
  const float* bg   = (const float*)d_in[6];
  const float* Wo   = (const float*)d_in[7];
  const float* bo   = (const float*)d_in[8];
  float* out = (float*)d_out;

  // workspace carve-up (f16), ~186 MB total
  _Float16* XN  = (_Float16*)d_ws;                      // [32768][256]
  _Float16* WT  = XN  + (size_t)N_TOK * D_MODEL;        // [2048][256]  Wq|Wk|Wv|Wg, [out][in]
  _Float16* WoT = WT  + (size_t)2048 * D_MODEL;         // [256][512]
  _Float16* Qb  = WoT + (size_t)D_MODEL * INNER_D;      // [bh][w][64]
  _Float16* Kb  = Qb  + (size_t)N_TOK * INNER_D;        // [bh][w][64]
  _Float16* Vt  = Kb  + (size_t)N_TOK * INNER_D;        // [bh][64][w]  (transposed)
  _Float16* Gb  = Vt  + (size_t)N_TOK * INNER_D;        // [t][512] sigmoid gates
  _Float16* AG  = Gb  + (size_t)N_TOK * INNER_D;        // [t][512] gated attn out

  ln_f16_kernel  <<<N_TOK, 32, 0, stream>>>(x, ln_g, ln_b, XN);
  cvt_wqkvg_kernel<<<2048, 256, 0, stream>>>(Wq, Wkv, Wg, WT);
  cvt_wo_kernel  <<<256, 256, 0, stream>>>(Wo, WoT);
  proj_kernel    <<<dim3(N_TOK / 16, 16), 32, 0, stream>>>(XN, WT, bg, Qb, Kb, Vt, Gb);
  attn_kernel    <<<dim3(SEQ / 16, NBATCH * N_HEADS), 32, 0, stream>>>(Qb, Kb, Vt, Gb, AG);
  out_kernel     <<<dim3(N_TOK / 16, 2), 32, 0, stream>>>(AG, WoT, bo, out);
}
